// TemporalGraphTransformer_21534966022825
// MI455X (gfx1250) — compile-verified, matrix operand order checked by reading, not run on hardware
//
#include <hip/hip_runtime.h>
#include <math.h>

// ---------------------------------------------------------------------------
// Problem constants (match reference)
// ---------------------------------------------------------------------------
#define BB   32
#define NN   512
#define INDIM 64
#define DD   256
#define HH   8
#define FF   1024
#define LL   3
#define PP   512
#define HDD  32
#define BN   (BB * NN)   // 16384 token rows
#define KMAX 1024        // largest GEMM K (FFN down-projection)

// ---------------------------------------------------------------------------
// CDNA5 WMMA types / helpers (wave32, 16x16x32 bf16 -> f32 accumulate)
// ---------------------------------------------------------------------------
typedef __attribute__((ext_vector_type(16))) __bf16 v16bf;
typedef __attribute__((ext_vector_type(8)))  __bf16 v8bf;
typedef __attribute__((ext_vector_type(8)))  float  v8f;

__device__ __forceinline__ v8f wmma_bf16(v16bf a, v16bf b, v8f c) {
    // (neg_a, A, neg_b, B, c_mod, C, reuse_a, reuse_b)
    return __builtin_amdgcn_wmma_f32_16x16x32_bf16(false, a, false, b,
                                                   (short)0, c, false, false);
}

// Load a 16x32 bf16 fragment (A layout; identical gather works for B = W^T
// when W is row-major [out, in]).  Per ISA 7.12.2 (wave32):
//   lane<16 : row=lane,    VGPR0..3 hold K=0..7,  VGPR4..7 hold K=16..23
//   lane>=16: row=lane-16, VGPR0..3 hold K=8..15, VGPR4..7 hold K=24..31
// => per lane: two contiguous 16-byte chunks at col kb and kb+16.
__device__ __forceinline__ v16bf load_frag(const __bf16* base, int ld) {
    int lane = threadIdx.x & 31;
    int r    = lane & 15;
    int kb   = (lane >> 4) * 8;
    const __bf16* p = base + (long)r * ld + kb;
    v8bf lo = *reinterpret_cast<const v8bf*>(p);
    v8bf hi = *reinterpret_cast<const v8bf*>(p + 16);
    v16bf f;
#pragma unroll
    for (int j = 0; j < 8; ++j) { f[j] = lo[j]; f[8 + j] = hi[j]; }
    return f;
}

// ---------------------------------------------------------------------------
// Kernel: temporal rank = inverse permutation of argsort (stable tie-break)
// ---------------------------------------------------------------------------
__global__ void rank_kernel(const float* __restrict__ tw, int* __restrict__ rank) {
    __shared__ float s[NN];
    int b = blockIdx.x;
    int i = threadIdx.x;
    s[i] = tw[(long)b * NN + i];
    __syncthreads();
    float vi = s[i];
    int cnt = 0;
#pragma unroll 8
    for (int j = 0; j < NN; ++j) {
        float vj = s[j];
        cnt += (vj < vi) || (vj == vi && j < i);
    }
    rank[(long)b * NN + i] = cnt < (PP - 1) ? cnt : (PP - 1);
}

// ---------------------------------------------------------------------------
// Kernel: f32 -> bf16 elementwise
// ---------------------------------------------------------------------------
__global__ void f2bf_kernel(const float* __restrict__ in, __bf16* __restrict__ out, long n) {
    long i = (long)blockIdx.x * blockDim.x + threadIdx.x;
    if (i < n) out[i] = (__bf16)in[i];
}

// ---------------------------------------------------------------------------
// Kernel: generic bf16 WMMA GEMM   C[M,Nc] = A[M,K] @ W[Nc,K]^T + bias
//
// One wave per 16x16 output tile; 8 waves (256 thr) per block share one M-tile.
// The shared 16xK activation strip (contiguous rows => contiguous bytes) is
// staged into LDS ONCE per block via the gfx1250 async global->LDS path
// (ASYNCcnt-tracked, drained with s_wait_asynccnt), then all 8 waves read
// their WMMA A-fragments from LDS; weight B-fragments stream from L2.
//
// EPI: 0 = store f32
//      1 = +pe[rank[row]] gather; store f32 AND bf16 (projection epilogue)
//      2 = exact GELU; store bf16 only (FFN up-projection)
// ---------------------------------------------------------------------------
template <int EPI>
__global__ __launch_bounds__(256) void gemm_kernel(
        const __bf16* __restrict__ A, const __bf16* __restrict__ W,
        const float* __restrict__ bias,
        float* __restrict__ Cf, __bf16* __restrict__ Cb,
        int K, int Nc,
        const int* __restrict__ rank, const float* __restrict__ pe) {
    __shared__ __bf16 Asmem[16 * KMAX];   // 32 KB max (K=1024)

    int wave = threadIdx.x >> 5;
    int lane = threadIdx.x & 31;
    long mt  = blockIdx.x;
    int  nt  = blockIdx.y * 8 + wave;

    const __bf16* Ab = A + mt * 16 * (long)K;   // 16 contiguous rows
    const __bf16* Wb = W + (long)nt * 16 * K;

    // --- async copy of the 16xK strip (16*K*2 bytes) into LDS -------------
    {
        unsigned bytes   = 16u * (unsigned)K * 2u;
        unsigned ldsBase = (unsigned)(unsigned long long)(uintptr_t)Asmem;
        for (unsigned off = threadIdx.x * 16u; off < bytes; off += 256u * 16u) {
            unsigned ldsAddr = ldsBase + off;
            const char* gp = (const char*)Ab + off;
            asm volatile("global_load_async_to_lds_b128 %0, %1, off"
                         :: "v"(ldsAddr), "v"(gp) : "memory");
        }
        asm volatile("s_wait_asynccnt 0" ::: "memory");
    }
    __syncthreads();

    v8f acc = {};
    for (int kk = 0; kk < K; kk += 32) {
        v16bf a = load_frag(Asmem + kk, K);   // ds_load_b128 x2
        v16bf b = load_frag(Wb + kk, K);      // global_load_b128 x2
        acc = wmma_bf16(a, b, acc);
    }

    int  col   = nt * 16 + (lane & 15);
    long mbase = mt * 16 + (lane >> 4) * 8;
    float bv = bias ? bias[col] : 0.0f;
#pragma unroll
    for (int i = 0; i < 8; ++i) {
        long row = mbase + i;
        float v  = acc[i] + bv;
        if constexpr (EPI == 1) {
            v += pe[(long)rank[row] * Nc + col];
        }
        if constexpr (EPI == 2) {
            v = 0.5f * v * (1.0f + erff(v * 0.70710678118654752f));
            Cb[row * (long)Nc + col] = (__bf16)v;
        } else {
            Cf[row * (long)Nc + col] = v;
            if constexpr (EPI == 1) Cb[row * (long)Nc + col] = (__bf16)v;
        }
    }
}

// ---------------------------------------------------------------------------
// Kernel: split fused QKV (f32 [B,N,3D]) into bf16 q,k [B,H,N,HD] and vT [B,H,HD,N]
// ---------------------------------------------------------------------------
__global__ void qkv_split_kernel(const float* __restrict__ qkv,
                                 __bf16* __restrict__ q, __bf16* __restrict__ k,
                                 __bf16* __restrict__ vT) {
    long idx = (long)blockIdx.x * blockDim.x + threadIdx.x;
    long total = (long)BN * 3 * DD;
    if (idx >= total) return;
    int col = (int)(idx % (3 * DD));
    long bn = idx / (3 * DD);
    int n = (int)(bn % NN);
    int b = (int)(bn / NN);
    int which = col >> 8;          // 0=q,1=k,2=v
    int hc    = col & 255;
    int h     = hc >> 5;
    int d     = hc & 31;
    __bf16 val = (__bf16)qkv[idx];
    long bh = (long)b * HH + h;
    if (which == 0)      q[(bh * NN + n) * HDD + d]  = val;
    else if (which == 1) k[(bh * NN + n) * HDD + d]  = val;
    else                 vT[(bh * HDD + d) * NN + n] = val;
}

// ---------------------------------------------------------------------------
// Kernel: masked attention for one (b, h, 16-query strip) per wave.
//   scores (WMMA, 1 K-step since HD==32) -> LDS f32 -> softmax -> bf16 probs
//   -> attn @ V (WMMA, A-frags from LDS, B-frags from pre-transposed vT)
// ---------------------------------------------------------------------------
__global__ __launch_bounds__(32) void attn_kernel(
        const __bf16* __restrict__ qg, const __bf16* __restrict__ kg,
        const __bf16* __restrict__ vTg, const unsigned char* __restrict__ mask,
        __bf16* __restrict__ o, float scale) {
    __shared__ float  sS[16 * NN];   // 32 KB f32 scores
    __shared__ __bf16 sP[16 * NN];   // 16 KB bf16 probabilities

    int lane = threadIdx.x & 31;
    int qt = blockIdx.x, h = blockIdx.y, b = blockIdx.z;
    long bh = (long)b * HH + h;

    const __bf16* qp = qg + (bh * NN + qt * 16) * HDD;
    v16bf qa = load_frag(qp, HDD);

    int c        = lane & 15;
    int mrowbase = (lane >> 4) * 8;

    for (int kt = 0; kt < NN / 16; ++kt) {
        const __bf16* kp = kg + (bh * NN + kt * 16) * HDD;
        v16bf kb = load_frag(kp, HDD);
        v8f s = {};
        s = wmma_bf16(qa, kb, s);
        int kcol = kt * 16 + c;
#pragma unroll
        for (int i = 0; i < 8; ++i) {
            int m    = mrowbase + i;
            int qrow = qt * 16 + m;
            float val = s[i] * scale;
            unsigned char mk = mask[((long)b * NN + qrow) * NN + kcol];
            val += mk ? 0.0f : -1e9f;
            sS[m * NN + kcol] = val;
        }
    }
    __syncthreads();

    // softmax over 512 keys: 2 lanes per row (16 rows x 256 cols each)
    {
        int row  = lane >> 1;
        int half = lane & 1;
        const float* sr = sS + row * NN + half * 256;
        float mx = -3.4e38f;
        for (int j = 0; j < 256; ++j) mx = fmaxf(mx, sr[j]);
        mx = fmaxf(mx, __shfl_xor(mx, 1, 32));
        float sum = 0.0f;
        for (int j = 0; j < 256; ++j) sum += expf(sr[j] - mx);
        sum += __shfl_xor(sum, 1, 32);
        float inv = 1.0f / sum;
        __bf16* pr = sP + row * NN + half * 256;
        for (int j = 0; j < 256; ++j) pr[j] = (__bf16)(expf(sr[j] - mx) * inv);
    }
    __syncthreads();

    // out[16, 32] = probs[16, 512] @ V[512, 32]; two 16-col N tiles (d 0..15, 16..31)
    v8f a0 = {}, a1 = {};
    for (int kt = 0; kt < NN / 32; ++kt) {
        v16bf af = load_frag(sP + kt * 32, NN);
        v16bf b0 = load_frag(vTg + (bh * HDD +  0) * NN + kt * 32, NN);
        v16bf b1 = load_frag(vTg + (bh * HDD + 16) * NN + kt * 32, NN);
        a0 = wmma_bf16(af, b0, a0);
        a1 = wmma_bf16(af, b1, a1);
    }
#pragma unroll
    for (int i = 0; i < 8; ++i) {
        int m = mrowbase + i;
        long row = (long)b * NN + qt * 16 + m;
        o[row * DD + h * HDD + c]      = (__bf16)a0[i];
        o[row * DD + h * HDD + 16 + c] = (__bf16)a1[i];
    }
}

// ---------------------------------------------------------------------------
// Kernel: h = LayerNorm(res + delta) ; emits f32 canonical + bf16 mirror.
// One wave per D=256 row (8 elems/lane), 8 rows per 256-thread block.
// ---------------------------------------------------------------------------
__global__ __launch_bounds__(256) void resln_kernel(
        const float* __restrict__ res, const float* __restrict__ delta,
        const float* __restrict__ g, const float* __restrict__ bvec,
        float* __restrict__ out_f, __bf16* __restrict__ out_b) {
    int wave = threadIdx.x >> 5;
    int lane = threadIdx.x & 31;
    long row = (long)blockIdx.x * 8 + wave;
    const float* r = res   + row * DD;
    const float* d = delta + row * DD;

    float x[8];
    float sum = 0.0f;
#pragma unroll
    for (int j = 0; j < 8; ++j) {
        int c = lane + 32 * j;
        x[j] = r[c] + d[c];
        sum += x[j];
    }
#pragma unroll
    for (int off = 16; off > 0; off >>= 1) sum += __shfl_xor(sum, off, 32);
    float mean = sum * (1.0f / DD);
    float var = 0.0f;
#pragma unroll
    for (int j = 0; j < 8; ++j) { float t = x[j] - mean; var += t * t; }
#pragma unroll
    for (int off = 16; off > 0; off >>= 1) var += __shfl_xor(var, off, 32);
    var *= (1.0f / DD);
    float inv = rsqrtf(var + 1e-5f);
#pragma unroll
    for (int j = 0; j < 8; ++j) {
        int c = lane + 32 * j;
        float y = (x[j] - mean) * inv * g[c] + bvec[c];
        out_f[row * DD + c] = y;
        out_b[row * DD + c] = (__bf16)y;
    }
}

// ---------------------------------------------------------------------------
// Host launcher
// ---------------------------------------------------------------------------
extern "C" void kernel_launch(void* const* d_in, const int* in_sizes, int n_in,
                              void* d_out, int out_size, void* d_ws, size_t ws_size,
                              hipStream_t stream) {
    const float* x     = (const float*)d_in[0];
    const float* tw    = (const float*)d_in[1];
    const unsigned char* adj = (const unsigned char*)d_in[2];  // bool array
    const float* Wp    = (const float*)d_in[3];
    const float* bp    = (const float*)d_in[4];
    const float* pe    = (const float*)d_in[5];
    const float* Wqkv  = (const float*)d_in[6];
    const float* bqkv  = (const float*)d_in[7];
    const float* Wo    = (const float*)d_in[8];
    const float* bo    = (const float*)d_in[9];
    const float* ln1g  = (const float*)d_in[10];
    const float* ln1b  = (const float*)d_in[11];
    const float* W1    = (const float*)d_in[12];
    const float* b1    = (const float*)d_in[13];
    const float* W2    = (const float*)d_in[14];
    const float* b2    = (const float*)d_in[15];
    const float* ln2g  = (const float*)d_in[16];
    const float* ln2b  = (const float*)d_in[17];
    float* out = (float*)d_out;

    // bump allocator over workspace (256-byte aligned)
    char* ws = (char*)d_ws;
    size_t cur = 0;
    auto alloc = [&](size_t bytes) -> void* {
        cur = (cur + 255) & ~(size_t)255;
        void* p = ws + cur;
        cur += bytes;
        return p;
    };

    int*    rank_p  = (int*)   alloc((size_t)BN * 4);
    __bf16* x_bf    = (__bf16*)alloc((size_t)BN * INDIM * 2);
    __bf16* Wp_bf   = (__bf16*)alloc((size_t)DD * INDIM * 2);
    __bf16* Wqkv_bf = (__bf16*)alloc((size_t)LL * 3 * DD * DD * 2);
    __bf16* Wo_bf   = (__bf16*)alloc((size_t)LL * DD * DD * 2);
    __bf16* W1_bf   = (__bf16*)alloc((size_t)LL * FF * DD * 2);
    __bf16* W2_bf   = (__bf16*)alloc((size_t)LL * DD * FF * 2);
    float*  h_f     = (float*) alloc((size_t)BN * DD * 4);
    __bf16* h_bf    = (__bf16*)alloc((size_t)BN * DD * 2);
    // big region: qkv_f32, later reused by ff1_bf + ff2_f (lifetimes disjoint)
    char*   bigA    = (char*)  alloc((size_t)BN * 3 * DD * 4);
    float*  qkv_f   = (float*) bigA;
    __bf16* ff1_bf  = (__bf16*)bigA;
    float*  ff2_f   = (float*)(bigA + (size_t)BN * FF * 2);
    // q/k region reused by oproj (attention consumes q/k before Wo GEMM runs)
    char*   bigB    = (char*)  alloc((size_t)BB * HH * NN * HDD * 2 * 2);
    __bf16* q_bf    = (__bf16*)bigB;
    __bf16* k_bf    = (__bf16*)(bigB + (size_t)BB * HH * NN * HDD * 2);
    float*  oproj_f = (float*) bigB;
    __bf16* vT_bf   = (__bf16*)alloc((size_t)BB * HH * HDD * NN * 2);
    __bf16* o_bf    = (__bf16*)alloc((size_t)BN * DD * 2);

    const float scale = 1.0f / sqrtf((float)HDD);

    // temporal ranks
    rank_kernel<<<BB, NN, 0, stream>>>(tw, rank_p);

    // f32 -> bf16 conversions (activations input + all weights)
    auto cvt = [&](const float* src, __bf16* dst, long n) {
        f2bf_kernel<<<(unsigned)((n + 255) / 256), 256, 0, stream>>>(src, dst, n);
    };
    cvt(x,    x_bf,    (long)BN * INDIM);
    cvt(Wp,   Wp_bf,   (long)DD * INDIM);
    cvt(Wqkv, Wqkv_bf, (long)LL * 3 * DD * DD);
    cvt(Wo,   Wo_bf,   (long)LL * DD * DD);
    cvt(W1,   W1_bf,   (long)LL * FF * DD);
    cvt(W2,   W2_bf,   (long)LL * DD * FF);

    // h = x @ Wp^T + bp + pe[rank]   (fp32 + bf16 mirror)
    gemm_kernel<1><<<dim3(BN / 16, DD / 128), 256, 0, stream>>>(
        x_bf, Wp_bf, bp, h_f, h_bf, INDIM, DD, rank_p, pe);

    for (int l = 0; l < LL; ++l) {
        // qkv = h @ Wqkv^T + bqkv
        gemm_kernel<0><<<dim3(BN / 16, (3 * DD) / 128), 256, 0, stream>>>(
            h_bf, Wqkv_bf + (size_t)l * 3 * DD * DD, bqkv + (size_t)l * 3 * DD,
            qkv_f, nullptr, DD, 3 * DD, nullptr, nullptr);

        // split into q,k [B,H,N,HD] and vT [B,H,HD,N] bf16
        {
            long total = (long)BN * 3 * DD;
            qkv_split_kernel<<<(unsigned)((total + 255) / 256), 256, 0, stream>>>(
                qkv_f, q_bf, k_bf, vT_bf);
        }

        // masked multi-head attention -> o_bf [B,N,D]
        attn_kernel<<<dim3(NN / 16, HH, BB), 32, 0, stream>>>(
            q_bf, k_bf, vT_bf, adj, o_bf, scale);

        // o @ Wo^T + bo
        gemm_kernel<0><<<dim3(BN / 16, DD / 128), 256, 0, stream>>>(
            o_bf, Wo_bf + (size_t)l * DD * DD, bo + (size_t)l * DD,
            oproj_f, nullptr, DD, DD, nullptr, nullptr);

        // h = LN(h + o_proj)
        resln_kernel<<<BN / 8, 256, 0, stream>>>(
            h_f, oproj_f, ln1g + (size_t)l * DD, ln1b + (size_t)l * DD, h_f, h_bf);

        // ff1 = gelu(h @ W1^T + b1)  (bf16)
        gemm_kernel<2><<<dim3(BN / 16, FF / 128), 256, 0, stream>>>(
            h_bf, W1_bf + (size_t)l * FF * DD, b1 + (size_t)l * FF,
            nullptr, ff1_bf, DD, FF, nullptr, nullptr);

        // ff2 = ff1 @ W2^T + b2  (f32)
        gemm_kernel<0><<<dim3(BN / 16, DD / 128), 256, 0, stream>>>(
            ff1_bf, W2_bf + (size_t)l * DD * FF, b2 + (size_t)l * DD,
            ff2_f, nullptr, FF, DD, nullptr, nullptr);

        // h = LN(h + ff2); final layer writes straight to d_out (fp32)
        float* dstf = (l == LL - 1) ? out : h_f;
        resln_kernel<<<BN / 8, 256, 0, stream>>>(
            h_f, ff2_f, ln2g + (size_t)l * DD, ln2b + (size_t)l * DD, dstf, h_bf);
    }
}
